// GNN_61314953118559
// MI455X (gfx1250) — compile-verified, matrix operand order checked by reading, not run on hardware
//
#include <hip/hip_runtime.h>
#include <hip/hip_bf16.h>
#include <stdint.h>

// ---------------------------------------------------------------------------
// GIN-style GNN forward for MI455X (gfx1250, wave32, WMMA).
//   h = x@Win+b ; 5x { agg=scatter_add ; z=agg@W1+b1 ; BN ; ReLU ; h=z@W2+b2 ; ReLU* }
// GEMMs: bf16 fragments, f32 accumulate via v_wmma_f32_16x16x32_bf16;
//        B staged in LDS per block, A ping-pong double-buffered from global
//        using two named registers (no indexed locals -> no scratch spills).
// Scatter: f32 global atomics (memory-bound stage, ~2GB/layer @ 23.3TB/s HBM).
// ---------------------------------------------------------------------------

typedef __attribute__((ext_vector_type(16))) __bf16 v16bf;
typedef __attribute__((ext_vector_type(8)))  __bf16 v8bf;
typedef __attribute__((ext_vector_type(8)))  float  v8f;

static inline int cdiv_i(int a, int b) { return (a + b - 1) / b; }

#define GEMM_KTC 10   // K-tiles staged in LDS per phase (4 nt * 10 kt * 1KB = 40KB)

// ----------------------------- small utility kernels -----------------------

__global__ void gnn_zero_f32(float* __restrict__ p, int n) {
  int i = blockIdx.x * blockDim.x + threadIdx.x;
  if (i < n) p[i] = 0.0f;
}

__global__ void gnn_copy_f4(const float4* __restrict__ src, float4* __restrict__ dst, int n4) {
  int i = blockIdx.x * blockDim.x + threadIdx.x;
  if (i < n4) dst[i] = src[i];
}

// fp32 [M,K] row-major -> bf16 [M,KP] row-major, zero padded (A operand).
__global__ void gnn_pack_a_bf16(const float* __restrict__ A, __bf16* __restrict__ Ab,
                                int K, int KP, int total) {
  int idx = blockIdx.x * blockDim.x + threadIdx.x;
  if (idx >= total) return;
  int c = idx % KP;
  int r = idx / KP;
  float v = (c < K) ? A[(size_t)r * K + c] : 0.0f;
  Ab[idx] = (__bf16)v;
}

// fp32 weight [K,N] row-major -> bf16 WMMA B fragments [nt16][kt][lane][16].
// Lane layout mirrors the ISA 16-bit A fragment: lane holds col n = nt*16+(lane&15);
// element e maps to k = kt*32 + (lane<16?0:8) + (e<8 ? e : 16+(e-8)).
__global__ void gnn_pack_b_frag(const float* __restrict__ W, __bf16* __restrict__ Bp,
                                int K, int N, int KT, int total) {
  int idx = blockIdx.x * blockDim.x + threadIdx.x;
  if (idx >= total) return;
  int e    = idx & 15;
  int lane = (idx >> 4) & 31;
  int rest = idx >> 9;
  int kt   = rest % KT;
  int nt   = rest / KT;
  int kl   = ((lane < 16) ? 0 : 8) + ((e < 8) ? e : 16 + (e - 8));
  int k    = kt * 32 + kl;
  int n    = nt * 16 + (lane & 15);
  float v  = (k < K && n < N) ? W[(size_t)k * N + n] : 0.0f;
  Bp[idx]  = (__bf16)v;
}

// ----------------------------- WMMA GEMM helpers ---------------------------

__device__ __forceinline__ v16bf gnn_load_a(const __bf16* ap) {
  union { v16bf v; v8bf h[2]; } a;
  a.h[0] = *(const v8bf*)(ap);        // K base0 + 0..7   (global_load_b128)
  a.h[1] = *(const v8bf*)(ap + 16);   // K base0 + 16..23 (global_load_b128)
  return a.v;
}

// One K-tile: 4 B fragments from LDS + 4 WMMAs into acc[0..3] (const indices).
__device__ __forceinline__ void gnn_wmma_step(const __bf16* sB, int kt, int lane,
                                              v16bf av, v8f acc[4]) {
  const __bf16* sbl = sB + ((size_t)(kt * 32 + lane) * 16);
#pragma unroll
  for (int j = 0; j < 4; ++j) {
    union { v16bf v; v8bf h[2]; } b;
    const __bf16* bp = sbl + (size_t)j * (GEMM_KTC * 512);
    b.h[0] = *(const v8bf*)(bp);       // ds_load_b128
    b.h[1] = *(const v8bf*)(bp + 8);   // ds_load_b128
    acc[j] = __builtin_amdgcn_wmma_f32_16x16x32_bf16(
        false, av, false, b.v, (short)0, acc[j], false, false);
  }
}

// ----------------------------- WMMA GEMM -----------------------------------
// C[M,Nreal] = Ab[M,KP](bf16) * Bfrag + bias, optional ReLU.
// Block = 8 waves: covers 128 (M) x 64 (N). Per wave: 16x64 tile, 4 accumulators.
// B super-tile (4 n-tiles x GEMM_KTC k-tiles) staged in LDS per phase;
// A fragments ping-pong in two named registers, loaded one K-tile ahead.
__global__ __launch_bounds__(256)
void gnn_gemm_wmma_bf16(const __bf16* __restrict__ Ab, const __bf16* __restrict__ Bp,
                        const float* __restrict__ bias, float* __restrict__ C,
                        int Mtiles, int MB8, int Nreal, int KT, int relu) {
  __shared__ __align__(16) __bf16 sB[4 * GEMM_KTC * 512];   // 40 KB

  const int lane = threadIdx.x & 31;
  const int wib  = threadIdx.x >> 5;
  const int mb   = blockIdx.x % MB8;
  const int ns   = blockIdx.x / MB8;
  const int nt0  = ns * 4;

  const int  mt     = mb * 8 + wib;
  const bool active = (mt < Mtiles);
  const int  KP     = KT * 32;
  const int  m0     = active ? mt * 16 : 0;   // inactive waves read row 0 (safe)
  const int  base0  = (lane < 16) ? 0 : 8;    // ISA 16-bit A fragment K layout
  const __bf16* __restrict__ arow = Ab + (size_t)(m0 + (lane & 15)) * KP + base0;

  v8f acc[4] = {v8f{}, v8f{}, v8f{}, v8f{}};

  for (int k0 = 0; k0 < KT; k0 += GEMM_KTC) {
    const int ktc = (KT - k0 < GEMM_KTC) ? (KT - k0) : GEMM_KTC;

    // ---- cooperative LDS fill: 4 n-tile segments, strided uint4 copy ----
    __syncthreads();   // previous phase's readers done before overwrite
    {
      const int seg = ktc * 64;   // uint4 per n-tile segment (<= 640)
#pragma unroll
      for (int j = 0; j < 4; ++j) {
        const uint4* __restrict__ g =
            (const uint4*)Bp + ((size_t)(nt0 + j) * KT + k0) * 64;
        uint4* s = (uint4*)sB + j * (GEMM_KTC * 64);
        for (int i = threadIdx.x; i < seg; i += 256) s[i] = g[i];
      }
    }
    __syncthreads();

    // ---- compute this K chunk; explicit 2-deep A pipeline, no indexed locals
    v16bf a0 = gnn_load_a(arow + (size_t)k0 * 32);
    int kt = 0;
    for (; kt < ktc - 1; kt += 2) {
      v16bf a1 = gnn_load_a(arow + (size_t)(k0 + kt + 1) * 32);
      gnn_wmma_step(sB, kt, lane, a0, acc);
      if (kt + 2 < ktc) a0 = gnn_load_a(arow + (size_t)(k0 + kt + 2) * 32);
      gnn_wmma_step(sB, kt + 1, lane, a1, acc);
    }
    if (ktc & 1) gnn_wmma_step(sB, ktc - 1, lane, a0, acc);
  }

  if (!active) return;

  // C/D fragment: VGPR r -> M = m0 + r + (lane>=16 ? 8 : 0), N = (lane&15).
  const int mrow = m0 + ((lane >> 4) << 3);
#pragma unroll
  for (int j = 0; j < 4; ++j) {
    const int n = (nt0 + j) * 16 + (lane & 15);
    if (n < Nreal) {
      const float bv = bias[n];
      float* cp = C + (size_t)mrow * Nreal + n;
#pragma unroll
      for (int r = 0; r < 8; ++r) {
        float v = acc[j][r] + bv;
        if (relu) v = fmaxf(v, 0.0f);
        cp[(size_t)r * Nreal] = v;
      }
    }
  }
}

// ----------------------------- scatter-add ---------------------------------
// agg[dst] += h[src] for each edge; agg pre-initialized to h (self loops).
__global__ void gnn_scatter_add(const float* __restrict__ h,
                                const long long* __restrict__ ei,
                                float* __restrict__ agg, int E, int total) {
  unsigned idx = blockIdx.x * 256u + threadIdx.x;
  if (idx >= (unsigned)total) return;
  unsigned e = idx / 300u;
  unsigned c = idx - e * 300u;
  long long s = ei[e];
  long long d = ei[(size_t)E + e];
  atomicAdd(agg + (size_t)d * 300 + c, h[(size_t)s * 300 + c]);
}

// ----------------------------- batch norm ----------------------------------
// Column-parallel partial sums: thread = column, blockIdx.y = row chunk.
__global__ void gnn_bn_stats(const float* __restrict__ z, float* __restrict__ sums,
                             int M, int H, int chunk) {
  int c = blockIdx.x * blockDim.x + threadIdx.x;
  if (c >= H) return;
  int r0 = blockIdx.y * chunk;
  int r1 = r0 + chunk; if (r1 > M) r1 = M;
  float s = 0.0f, q = 0.0f;
  for (int r = r0; r < r1; ++r) {
    float v = z[(size_t)r * H + c];   // consecutive threads -> contiguous 128B
    s += v; q += v * v;
  }
  atomicAdd(&sums[c], s);
  atomicAdd(&sums[H + c], q);
}

__global__ void gnn_bn_coef(const float* __restrict__ sums, const float* __restrict__ gamma,
                            const float* __restrict__ beta, float* __restrict__ coef,
                            int M, int H) {
  int c = blockIdx.x * blockDim.x + threadIdx.x;
  if (c >= H) return;
  float inv = 1.0f / (float)M;
  float mu  = sums[c] * inv;
  float var = sums[H + c] * inv - mu * mu;
  float rs  = rsqrtf(var + 1e-5f);
  float sc  = gamma[c] * rs;
  coef[c]     = sc;                 // scale
  coef[H + c] = beta[c] - mu * sc;  // shift
}

// z -> relu(z*scale+shift) -> bf16, zero padded to HP (next GEMM A operand).
__global__ void gnn_bn_apply_pack(const float* __restrict__ z, const float* __restrict__ coef,
                                  __bf16* __restrict__ zb, int H, int HP, int total) {
  int idx = blockIdx.x * blockDim.x + threadIdx.x;
  if (idx >= total) return;
  int c = idx % HP;
  int r = idx / HP;
  float v = 0.0f;
  if (c < H) {
    v = z[(size_t)r * H + c] * coef[c] + coef[H + c];
    v = fmaxf(v, 0.0f);
  }
  zb[idx] = (__bf16)v;
}

// ----------------------------- launcher ------------------------------------

extern "C" void kernel_launch(void* const* d_in, const int* in_sizes, int n_in,
                              void* d_out, int out_size, void* d_ws, size_t ws_size,
                              hipStream_t stream) {
  const float*     x     = (const float*)d_in[0];
  const long long* ei    = (const long long*)d_in[1];   // reference dtype: int64
  // d_in[2] edge_attr: unused by the reference computation
  const float*     lin_w = (const float*)d_in[3];
  const float*     lin_b = (const float*)d_in[4];
  const float*     w1    = (const float*)d_in[5];
  const float*     b1    = (const float*)d_in[6];
  const float*     gamma = (const float*)d_in[7];
  const float*     beta  = (const float*)d_in[8];
  const float*     w2    = (const float*)d_in[9];
  const float*     b2    = (const float*)d_in[10];

  const int IN  = 232, EMB = 300, HID = 600, L = 5;
  const int N   = in_sizes[0] / IN;        // 50000 (= 3125 * 16)
  const int E   = in_sizes[1] / 2;         // 800000
  const int MT  = N / 16;                  // M tiles
  const int MB8 = cdiv_i(MT, 8);           // 8-Mtile blocks

  // K pads (x32) and N layout (16-tiles, 64-wide super-tiles)
  const int K1P = 256, KT1 = 8,  NT1 = 20, NS1 = 5;    // 232->256, N 300->320
  const int K2P = 320, KT2 = 10, NT2 = 40, NS2 = 10;   // 300->320, N 600->640
  const int K3P = 608, KT3 = 19, NT3 = 20, NS3 = 5;    // 600->608, N 300->320

  // Workspace layout (bytes, 256-aligned). A-operand buffer reused per stage.
  char* base = (char*)d_ws;
  const size_t SZ_ABUF = (size_t)N * K3P * 2;              // 60.8 MB (max of 3 stages)
  const size_t SZ_H    = (size_t)N * EMB * 4;              // 60 MB
  const size_t SZ_Z    = (size_t)N * HID * 4;              // 120 MB
  size_t off = 0;
  __bf16* abuf = (__bf16*)(base + off); off += SZ_ABUF;
  float*  h    = (float*)(base + off);  off += SZ_H;
  float*  agg  = (float*)(base + off);  off += SZ_H;
  float*  z    = (float*)(base + off);  off += SZ_Z;
  __bf16* bp   = (__bf16*)(base + off); off += 512000;     // >= max B-frag (409,600 B)
  float*  sums = (float*)(base + off);  off += 4864;       // 2*HID f32
  float*  coef = (float*)(base + off);  off += 4864;

  const int TPB = 256;

  // ---- input GEMM: h = x @ Win + b ----
  {
    int tot = N * K1P;
    gnn_pack_a_bf16<<<cdiv_i(tot, TPB), TPB, 0, stream>>>(x, abuf, IN, K1P, tot);
    int totb = NT1 * KT1 * 512;
    gnn_pack_b_frag<<<cdiv_i(totb, TPB), TPB, 0, stream>>>(lin_w, bp, IN, EMB, KT1, totb);
    gnn_gemm_wmma_bf16<<<MB8 * NS1, TPB, 0, stream>>>(
        abuf, bp, lin_b, h, MT, MB8, EMB, KT1, 0);
  }

  const int n4    = (N * EMB) / 4;
  const int stot  = E * 300;               // 240M, fits int
  const int chunk = cdiv_i(N, 64);

  for (int l = 0; l < L; ++l) {
    // ---- agg = h (self loops) + scatter-add over edges ----
    gnn_copy_f4<<<cdiv_i(n4, TPB), TPB, 0, stream>>>((const float4*)h, (float4*)agg, n4);
    gnn_scatter_add<<<cdiv_i(stot, TPB), TPB, 0, stream>>>(h, ei, agg, E, stot);

    // ---- z = agg @ W1[l] + b1[l] ----
    {
      int tot = N * K2P;
      gnn_pack_a_bf16<<<cdiv_i(tot, TPB), TPB, 0, stream>>>(agg, abuf, EMB, K2P, tot);
      int totb = NT2 * KT2 * 512;
      gnn_pack_b_frag<<<cdiv_i(totb, TPB), TPB, 0, stream>>>(
          w1 + (size_t)l * EMB * HID, bp, EMB, HID, KT2, totb);
      gnn_gemm_wmma_bf16<<<MB8 * NS2, TPB, 0, stream>>>(
          abuf, bp, b1 + (size_t)l * HID, z, MT, MB8, HID, KT2, 0);
    }

    // ---- BatchNorm (train-mode batch stats) + ReLU, pack bf16 ----
    gnn_zero_f32<<<cdiv_i(2 * HID, TPB), TPB, 0, stream>>>(sums, 2 * HID);
    {
      dim3 g(cdiv_i(HID, 128), 64);
      gnn_bn_stats<<<g, 128, 0, stream>>>(z, sums, N, HID, chunk);
    }
    gnn_bn_coef<<<cdiv_i(HID, TPB), TPB, 0, stream>>>(
        sums, gamma + (size_t)l * HID, beta + (size_t)l * HID, coef, N, HID);
    {
      int tot = N * K3P;
      gnn_bn_apply_pack<<<cdiv_i(tot, TPB), TPB, 0, stream>>>(z, coef, abuf, HID, K3P, tot);
    }

    // ---- h = z @ W2[l] + b2[l], ReLU between layers, last writes d_out ----
    {
      int totb = NT3 * KT3 * 512;
      gnn_pack_b_frag<<<cdiv_i(totb, TPB), TPB, 0, stream>>>(
          w2 + (size_t)l * HID * EMB, bp, HID, EMB, KT3, totb);
      float* out = (l == L - 1) ? (float*)d_out : h;
      gnn_gemm_wmma_bf16<<<MB8 * NS3, TPB, 0, stream>>>(
          abuf, bp, b2 + (size_t)l * EMB, out, MT, MB8, EMB, KT3, (l == L - 1) ? 0 : 1);
    }
  }
}
